// _SharedCodebookResidualQuantizer_18674517803204
// MI455X (gfx1250) — compile-verified, matrix operand order checked by reading, not run on hardware
//
#include <hip/hip_runtime.h>
#include <cstdint>

// ---------------------------------------------------------------------------
// Residual quantizer (4 levels) for MI455X / gfx1250.
//  - distance GEMM: v_wmma_f32_16x16x32_bf16 (bf16 scores feed only argmin)
//  - residual kept exact in fp32 LDS; bf16 shadow maintained for WMMA A
//  - codebook pre-converted to bf16 in workspace; chunks staged to LDS with
//    global_load_async_to_lds_b128 (ASYNCcnt) per CDNA5 ISA ch.10/15.18
//  - LDS layout puts the codebook chunk at offset 0 so the 8 B-tile
//    ds_load_b128 per d-step use 16-bit immediate offsets (no VALU adds)
// ---------------------------------------------------------------------------

typedef __attribute__((ext_vector_type(16))) __bf16 v16bf;
typedef __attribute__((ext_vector_type(8)))  float  v8f;

#define N_TOTAL 32768
#define DIM     512
#define KCB     1024
#define LEVELS  4
#define ROWS_WG 64            // rows per workgroup tile
#define THREADS 128           // 4 wave32
#define CHUNK   64            // codebook entries staged in LDS per iteration
#define NCHUNK  (KCB / CHUNK) // 16
#define DSTEPS  (DIM / 32)    // 16 WMMA k-steps of 32 bf16
#define KTILES  (CHUNK / 16)  // 4 accumulator tiles per chunk

// ---- LDS layout (dynamic shared) ----
// cbh first: B-tile addresses = lane_base + t*16384 + {0,32} -> DS immediates.
#define OFF_CBH     0                             // ushort [64*512]   65536 B
#define OFF_RESIDH  (OFF_CBH    + CHUNK*DIM*2)    // ushort [64*512]   65536 B
#define OFF_RESID   (OFF_RESIDH + ROWS_WG*DIM*2)  // float  [64*512]  131072 B
#define OFF_CBSQ    (OFF_RESID  + ROWS_WG*DIM*4)  // float  [1024]      4096 B
#define OFF_ROWIDX  (OFF_CBSQ   + KCB*4)          // int    [64]         256 B
#define OFF_RED     (OFF_ROWIDX + ROWS_WG*4)      // float  [128]        512 B
#define SMEM_BYTES  (OFF_RED    + THREADS*4)

union V16 {
  v16bf        v;
  unsigned int u[8];
};

// pack two fp32 into one dword of two bf16 (round-to-nearest-even-ish)
__device__ __forceinline__ unsigned int pk_bf16(float lo, float hi) {
  unsigned int a = __float_as_uint(lo);
  unsigned int b = __float_as_uint(hi);
  unsigned int ra = ((a + 0x7FFFu + ((a >> 16) & 1u)) >> 16) & 0x0000FFFFu;
  unsigned int rb =  (b + 0x7FFFu + ((b >> 16) & 1u))        & 0xFFFF0000u;
  return ra | rb;
}

// async global -> LDS copy of 16 bytes per lane (CDNA5, tracked on ASYNCcnt)
__device__ __forceinline__ void async_copy_b128(unsigned lds_off,
                                                unsigned long long gaddr) {
  asm volatile("global_load_async_to_lds_b128 %0, %1, off"
               :: "v"(lds_off), "v"(gaddr) : "memory");
}
__device__ __forceinline__ void async_wait0() {
  asm volatile("s_wait_asynccnt 0x0" ::: "memory");
}

// --------------------------- setup kernels ---------------------------------

__global__ void rq_init_kernel(float* closs, float* usage) {
  int i = blockIdx.x * blockDim.x + threadIdx.x;
  if (i == 0) *closs = 0.0f;
  if (i < KCB) usage[i] = 0.0f;
}

// per-entry ||c||^2 into workspace
__global__ void rq_cbsq_kernel(const float* __restrict__ cb, float* __restrict__ cbsq) {
  __shared__ float red[128];
  const int r = blockIdx.x;
  float s = 0.0f;
  for (int c = threadIdx.x; c < DIM; c += 128) {
    float v = cb[(size_t)r * DIM + c];
    s += v * v;
  }
  red[threadIdx.x] = s;
  __syncthreads();
  for (int o = 64; o > 0; o >>= 1) {
    if (threadIdx.x < o) red[threadIdx.x] += red[threadIdx.x + o];
    __syncthreads();
  }
  if (threadIdx.x == 0) cbsq[r] = red[0];
}

// codebook fp32 -> bf16 once, into workspace (row-major, packed pairs)
__global__ void rq_cbh_kernel(const float* __restrict__ cb,
                              unsigned int* __restrict__ cbh_g) {
  int i = blockIdx.x * blockDim.x + threadIdx.x;   // dword index
  if (i < KCB * DIM / 2)
    cbh_g[i] = pk_bf16(cb[(size_t)i * 2], cb[(size_t)i * 2 + 1]);
}

// ------------------------------ main kernel --------------------------------

__global__ void __launch_bounds__(THREADS, 1)
rq_main_kernel(const float* __restrict__ z, const float* __restrict__ cb,
               const float* __restrict__ cbsq_g,
               const unsigned int* __restrict__ cbh_g,
               float* __restrict__ out_zq, float* __restrict__ out_closs,
               float* __restrict__ out_codes, float* __restrict__ out_usage) {
  extern __shared__ char smem[];
  unsigned short* cbh    = (unsigned short*)(smem + OFF_CBH);
  unsigned short* residh = (unsigned short*)(smem + OFF_RESIDH);
  float*          resid  = (float*)(smem + OFF_RESID);
  float*          cbsq_l = (float*)(smem + OFF_CBSQ);
  int*            rowidx = (int*)(smem + OFF_ROWIDX);
  float*          red    = (float*)(smem + OFF_RED);

  const int tid  = threadIdx.x;
  const int wave = tid >> 5;
  const int lane = tid & 31;
  const int half = lane >> 4;        // 0: K{0..7,16..23}   1: K{8..15,24..31}
  const int l16  = lane & 15;
  const int wrow = wave * 16;        // wave's 16-row slice inside the tile
  const int arow = wrow + l16;       // this lane's A-matrix row
  const int rowbase = blockIdx.x * ROWS_WG;

  const unsigned lds_base = (unsigned)__builtin_amdgcn_groupstaticsize();

  // ---- async stage z tile into LDS residual (fp32, raw byte copy) ----
  {
    const unsigned long long gbase =
        (unsigned long long)(const void*)(z + (size_t)rowbase * DIM);
    #pragma unroll 4
    for (int i = 0; i < (ROWS_WG * DIM * 4 / 16) / THREADS; ++i) {
      unsigned v = (unsigned)(tid + i * THREADS);             // 16B units
      async_copy_b128(lds_base + OFF_RESID + v * 16u, gbase + v * 16ull);
    }
    // stage cbsq (4 KB of floats) synchronously meanwhile
    #pragma unroll
    for (int i = 0; i < KCB / THREADS; ++i)
      cbsq_l[tid + i * THREADS] = cbsq_g[tid + i * THREADS];
    async_wait0();
  }
  __syncthreads();

  // ---- build bf16 shadow of residual ----
  {
    const float4* r4 = (const float4*)resid;
    unsigned int* rh2 = (unsigned int*)residh;
    #pragma unroll 4
    for (int i = 0; i < (ROWS_WG * DIM / 4) / THREADS; ++i) {
      int v = tid + i * THREADS;
      float4 f = r4[v];
      rh2[v * 2 + 0] = pk_bf16(f.x, f.y);
      rh2[v * 2 + 1] = pk_bf16(f.z, f.w);
    }
  }
  __syncthreads();

  float sumsq = 0.0f;   // per-thread closs accumulator (raw sum of squares)

  for (int level = 0; level < LEVELS; ++level) {
    float bestd[8];
    int   bestk[8];
    #pragma unroll
    for (int i = 0; i < 8; ++i) { bestd[i] = 3.0e38f; bestk[i] = 0; }

    for (int c = 0; c < NCHUNK; ++c) {
      const int kbase = c * CHUNK;

      // ---- async stage pre-converted bf16 codebook chunk (64 KB raw) ----
      {
        const unsigned long long gbase =
            (unsigned long long)(const void*)(cbh_g + (size_t)kbase * (DIM / 2));
        #pragma unroll 4
        for (int i = 0; i < (CHUNK * DIM * 2 / 16) / THREADS; ++i) {
          unsigned v = (unsigned)(tid + i * THREADS);         // 16B units
          async_copy_b128(lds_base + OFF_CBH + v * 16u, gbase + v * 16ull);
        }
        if (c + 1 < NCHUNK)
          __builtin_prefetch(cbh_g + (size_t)(kbase + CHUNK) * (DIM / 2), 0, 1);
        async_wait0();
      }
      __syncthreads();

      // ---- score GEMM: 16 rows x 64 entries per wave via WMMA bf16 ----
      v8f acc[KTILES] = {};
      {
        // running lane pointers; all B-tile offsets become DS immediates
        const unsigned short* ap = residh + arow * DIM + half * 8;
        const unsigned short* bp = cbh + l16 * DIM + half * 8;
        #pragma unroll 2
        for (int ds2 = 0; ds2 < DSTEPS; ++ds2) {
          uint4 a0 = *(const uint4*)(ap);
          uint4 a1 = *(const uint4*)(ap + 16);
          V16 A;
          A.u[0] = a0.x; A.u[1] = a0.y; A.u[2] = a0.z; A.u[3] = a0.w;
          A.u[4] = a1.x; A.u[5] = a1.y; A.u[6] = a1.z; A.u[7] = a1.w;

          V16 B[KTILES];
          #pragma unroll
          for (int t = 0; t < KTILES; ++t) {
            uint4 b0 = *(const uint4*)(bp + t * 16 * DIM);       // +t*16384 B
            uint4 b1 = *(const uint4*)(bp + t * 16 * DIM + 16);  // +32 B
            B[t].u[0] = b0.x; B[t].u[1] = b0.y; B[t].u[2] = b0.z; B[t].u[3] = b0.w;
            B[t].u[4] = b1.x; B[t].u[5] = b1.y; B[t].u[6] = b1.z; B[t].u[7] = b1.w;
          }
          #pragma unroll
          for (int t = 0; t < KTILES; ++t)
            acc[t] = __builtin_amdgcn_wmma_f32_16x16x32_bf16(
                false, A.v, false, B[t].v, (short)0, acc[t], false, false);

          ap += 32;   // next 32 bf16 of the contraction dim
          bp += 32;
        }
      }

      // ---- fold into per-slot running argmin:  d = ||c||^2 - 2 r.c ----
      #pragma unroll
      for (int t = 0; t < KTILES; ++t) {
        const int n = kbase + t * 16 + l16;     // codebook index for this lane
        const float csq = cbsq_l[n];
        #pragma unroll
        for (int i = 0; i < 8; ++i) {
          float d = csq - 2.0f * acc[t][i];
          if (d < bestd[i]) { bestd[i] = d; bestk[i] = n; }
        }
      }
      __syncthreads();   // chunk LDS about to be overwritten
    }

    // ---- cross-lane argmin: butterfly within each 16-lane half ----
    #pragma unroll
    for (int m = 1; m < 16; m <<= 1) {
      #pragma unroll
      for (int i = 0; i < 8; ++i) {
        float od = __shfl_xor(bestd[i], m, 32);
        int   ok = __shfl_xor(bestk[i], m, 32);
        if (od < bestd[i] || (od == bestd[i] && ok < bestk[i])) {
          bestd[i] = od; bestk[i] = ok;
        }
      }
    }
    if (l16 == 0) {
      #pragma unroll
      for (int i = 0; i < 8; ++i)
        rowidx[wrow + half * 8 + i] = bestk[i];   // C layout: vgpr i -> row i / i+8
    }
    __syncthreads();

    // ---- residual update (fp32, exact) + bf16 shadow + stats ----
    {
      const int r = tid >> 1;
      const int colbase4 = (tid & 1) * (DIM / 8);   // 2 threads/row, 256 cols each
      const int idx = rowidx[r];
      const float4* q4 = (const float4*)(cb + (size_t)idx * DIM);
      float4* r4 = (float4*)(resid + r * DIM);
      unsigned int* rh2 = (unsigned int*)(residh + r * DIM);
      #pragma unroll 8
      for (int c4i = colbase4; c4i < colbase4 + DIM / 8; ++c4i) {
        float4 q = q4[c4i];
        float4 rr = r4[c4i];
        rr.x -= q.x; rr.y -= q.y; rr.z -= q.z; rr.w -= q.w;
        r4[c4i] = rr;
        rh2[c4i * 2 + 0] = pk_bf16(rr.x, rr.y);
        rh2[c4i * 2 + 1] = pk_bf16(rr.z, rr.w);
        sumsq += rr.x * rr.x + rr.y * rr.y + rr.z * rr.z + rr.w * rr.w;
      }
      if ((tid & 1) == 0) {
        out_codes[(size_t)(rowbase + r) * LEVELS + level] = (float)idx;
        atomicAdd(&out_usage[idx], 1.0f);
      }
    }
    __syncthreads();
  }

  // ---- z_quantized = z + sg(qsum - z)  ==  z - final_residual ----
  {
    const float4* z4 = (const float4*)(z + (size_t)rowbase * DIM);
    const float4* r4 = (const float4*)resid;
    float4* o4 = (float4*)(out_zq + (size_t)rowbase * DIM);
    #pragma unroll 4
    for (int i = 0; i < (ROWS_WG * DIM / 4) / THREADS; ++i) {
      int v = tid + i * THREADS;
      float4 a = z4[v], b = r4[v];
      float4 o;
      o.x = a.x - b.x; o.y = a.y - b.y; o.z = a.z - b.z; o.w = a.w - b.w;
      o4[v] = o;
    }
  }

  // ---- closs block reduction + global atomic ----
  red[tid] = sumsq;
  __syncthreads();
  for (int o = THREADS / 2; o > 0; o >>= 1) {
    if (tid < o) red[tid] += red[tid + o];
    __syncthreads();
  }
  if (tid == 0)
    atomicAdd(out_closs, red[0] * (1.0f / ((float)N_TOTAL * (float)DIM)));
}

// ------------------------------- launcher ----------------------------------

extern "C" void kernel_launch(void* const* d_in, const int* in_sizes, int n_in,
                              void* d_out, int out_size, void* d_ws, size_t ws_size,
                              hipStream_t stream) {
  const float* z  = (const float*)d_in[0];   // [32768, 512]
  const float* cb = (const float*)d_in[1];   // [1024, 512]

  float* out = (float*)d_out;
  // output layout (concatenated flat, reference return order):
  float* out_zq    = out;                                   // [N*D]
  float* out_closs = out + (size_t)N_TOTAL * DIM;           // [1]
  float* out_codes = out_closs + 1;                         // [N*LEVELS]
  float* out_usage = out_codes + (size_t)N_TOTAL * LEVELS;  // [K]

  // workspace: cbsq [1024 f32] then bf16 codebook [1024*512 bf16 = 1 MB]
  float*        cbsq  = (float*)d_ws;
  unsigned int* cbh_g = (unsigned int*)((char*)d_ws + 4096);

  rq_init_kernel<<<(KCB + 255) / 256, 256, 0, stream>>>(out_closs, out_usage);
  rq_cbsq_kernel<<<KCB, 128, 0, stream>>>(cb, cbsq);
  rq_cbh_kernel<<<(KCB * DIM / 2 + 255) / 256, 256, 0, stream>>>(cb, cbh_g);

  (void)hipFuncSetAttribute((const void*)rq_main_kernel,
                            hipFuncAttributeMaxDynamicSharedMemorySize,
                            SMEM_BYTES);
  rq_main_kernel<<<N_TOTAL / ROWS_WG, THREADS, SMEM_BYTES, stream>>>(
      z, cb, cbsq, cbh_g, out_zq, out_closs, out_codes, out_usage);

  (void)in_sizes; (void)n_in; (void)out_size; (void)ws_size;
}
